// AttentionPPO_7567732375716
// MI455X (gfx1250) — compile-verified, matrix operand order checked by reading, not run on hardware
//
#include <hip/hip_runtime.h>
#include <math.h>

// ---- CDNA5 wave32 WMMA types ------------------------------------------------
typedef __attribute__((ext_vector_type(16))) _Float16 v16h;
typedef __attribute__((ext_vector_type(8)))  _Float16 v8h;
typedef __attribute__((ext_vector_type(8)))  float    v8f;
typedef __attribute__((ext_vector_type(4)))  float    v4f;

#define DM    64      // d_model
#define NTOK  8       // tokens
#define TD    50      // token dim
#define NH    4       // heads
#define HD    16      // head dim
#define DFF   256     // ffn dim
#define WAVES 4       // waves per block
#define EPW   2       // batch elements per wave (16 rows = one WMMA M tile)

__device__ __forceinline__ v8f wmma_f16(v16h a, v16h b, v8f c) {
  // D = A(16x32 f16) * B(32x16 f16) + C(16x16 f32)
  return __builtin_amdgcn_wmma_f32_16x16x32_f16(false, a, false, b, (short)0, c,
                                                false, false);
}

// A-operand (16x32 f16) from LDS row-major [16][strideH] f16.
// ISA layout: lane m=l&15; hi=l>>4; elems 0..7 <- K=kb+hi*8+0..7,
//             elems 8..15 <- K=kb+16+hi*8+0..7. All 16B-aligned ds_load_b128.
__device__ __forceinline__ v16h loadA_lds(const _Float16* base, int strideH,
                                          int m, int hi, int kb) {
  const _Float16* p = base + m * strideH + kb + hi * 8;
  v8h lo = *(const v8h*)(p);
  v8h hh = *(const v8h*)(p + 16);
  return __builtin_shufflevector(lo, hh, 0,1,2,3,4,5,6,7,8,9,10,11,12,13,14,15);
}

// B-operand (32x16 f16) from global row-major W[N][K] f32 (K mult of 8).
// Lane holds column n; same K packing as A (mirrored). Aligned v4f loads + cvt.
__device__ __forceinline__ v16h loadB_gl(const float* W, int K, int n,
                                         int hi, int kb) {
  const float* p = W + (size_t)n * K + kb + hi * 8;
  v4f f0 = *(const v4f*)(p);
  v4f f1 = *(const v4f*)(p + 4);
  v4f f2 = *(const v4f*)(p + 16);
  v4f f3 = *(const v4f*)(p + 20);
  v16h r;
#pragma unroll
  for (int i = 0; i < 4; ++i) {
    r[i]      = (_Float16)f0[i];
    r[4 + i]  = (_Float16)f1[i];
    r[8 + i]  = (_Float16)f2[i];
    r[12 + i] = (_Float16)f3[i];
  }
  return r;
}

// B-operand for token projection: W[64][50] f32, K zero-padded to 64.
// Branchless: clamp index so the load is always in-bounds, then select 0
// (v_cndmask) for padded K. After unrolling only the straddling chunk keeps
// selects; no exec-mask branches.
__device__ __forceinline__ v16h loadB_tok(const float* W, int n, int hi, int kb) {
  const float* row = W + n * TD;
  v16h r;
#pragma unroll
  for (int i = 0; i < 8; ++i) {
    const int k0  = kb + hi * 8 + i;
    const int k1  = kb + 16 + hi * 8 + i;
    const int k0c = (k0 < TD) ? k0 : (TD - 1);
    const int k1c = (k1 < TD) ? k1 : (TD - 1);
    const float w0 = row[k0c];
    const float w1 = row[k1c];
    r[i]     = (_Float16)((k0 < TD) ? w0 : 0.0f);
    r[8 + i] = (_Float16)((k1 < TD) ? w1 : 0.0f);
  }
  return r;
}

// LayerNorm over 64 cols of fp32 LDS row -> f16 LDS (WMMA A staging).
// Lane handles (row=mA, half hi): 32 elems; partner-lane combine via shfl_xor(16).
__device__ __forceinline__ void ln_to_f16(const float* X, _Float16* A16,
                                          const float* lw, const float* lb,
                                          int mA, int hi) {
  float vals[32];
  float sum = 0.f, sq = 0.f;
  const float* row = X + mA * DM + hi * 32;
#pragma unroll
  for (int j = 0; j < 8; ++j) {
    v4f v = *(const v4f*)(row + 4 * j);
#pragma unroll
    for (int t = 0; t < 4; ++t) { vals[4*j+t] = v[t]; sum += v[t]; sq += v[t]*v[t]; }
  }
  sum += __shfl_xor(sum, 16);
  sq  += __shfl_xor(sq, 16);
  const float mu   = sum * (1.0f / 64.0f);
  const float var  = sq * (1.0f / 64.0f) - mu * mu;
  const float rstd = rsqrtf(var + 1e-5f);
#pragma unroll
  for (int j = 0; j < 32; ++j) {
    const int c = hi * 32 + j;
    A16[mA * DM + c] = (_Float16)((vals[j] - mu) * rstd * lw[c] + lb[c]);
  }
}

__global__ __launch_bounds__(WAVES * 32)
void ppo_transformer_fwd(
    const float* __restrict__ obs,
    const float* __restrict__ tokw, const float* __restrict__ tokb,
    const float* __restrict__ pos,
    const float* __restrict__ ln1w, const float* __restrict__ ln1b,
    const float* __restrict__ inw,  const float* __restrict__ inb,
    const float* __restrict__ outw, const float* __restrict__ outb,
    const float* __restrict__ ln2w, const float* __restrict__ ln2b,
    const float* __restrict__ f1w,  const float* __restrict__ f1b,
    const float* __restrict__ f2w,  const float* __restrict__ f2b,
    const float* __restrict__ a1w,  const float* __restrict__ a1b,
    const float* __restrict__ a2w,  const float* __restrict__ a2b,
    const float* __restrict__ c1w,  const float* __restrict__ c1b,
    const float* __restrict__ c2w,  const float* __restrict__ c2b,
    float* __restrict__ out_logits, float* __restrict__ out_value,
    float* __restrict__ out_attn)
{
  __shared__ __align__(16) float    sx[WAVES][16 * DM];     // residual x (fp32)
  __shared__ __align__(16) _Float16 sa[WAVES][16 * DM];     // xn / attn-out (f16)
  __shared__ __align__(16) _Float16 sbuf[WAVES][16 * DFF];  // tok / qkv / h (f16)

  const int tid = threadIdx.x;
  const int l   = tid & 31;   // lane in wave32
  const int w   = tid >> 5;   // wave id
  const int hi  = l >> 4;     // half-wave (K sub-block / C row group)
  const int nl  = l & 15;     // column within tile (B/C operand)
  const int mA  = l & 15;     // row within tile (A operand)

  float*    X   = sx[w];
  _Float16* A16 = sa[w];
  _Float16* Bf  = sbuf[w];

  const int e0 = blockIdx.x * (WAVES * EPW) + w * EPW;  // first batch elem

  // ---- stage zero-padded tokens (A for token projection) into Bf[16][64] ----
  {
    const int elem = e0 + (mA >> 3);
    const int s    = mA & 7;
    const float* orow = obs + (size_t)elem * 400 + (size_t)s * TD;
#pragma unroll
    for (int j = 0; j < 32; ++j) {
      const int c  = hi * 32 + j;
      const int cc = (c < TD) ? c : (TD - 1);
      const float v = orow[cc];
      Bf[mA * DM + c] = (_Float16)((c < TD) ? v : 0.0f);
    }
  }

  // ---- token projection + bias + positional embedding -> X (fp32 LDS) ----
#pragma unroll
  for (int nt = 0; nt < 4; ++nt) {
    const int n0 = nt * 16;
    v8f c = {0.f,0.f,0.f,0.f,0.f,0.f,0.f,0.f};
#pragma unroll
    for (int kb = 0; kb < DM; kb += 32) {
      v16h a = loadA_lds(Bf, DM, mA, hi, kb);
      v16h b = loadB_tok(tokw, n0 + nl, hi, kb);
      c = wmma_f16(a, b, c);
    }
    const float bias = tokb[n0 + nl];
#pragma unroll
    for (int r = 0; r < 8; ++r) {
      const int m = r + 8 * hi;
      X[m * DM + n0 + nl] = c[r] + bias + pos[(m & 7) * DM + n0 + nl];
    }
  }

  float probs0[EPW][NTOK];  // layer-0 attention probs (this lane's (h,q) triple)

  for (int L = 0; L < 2; ++L) {
    // ---- LN1 -> A16 ----
    ln_to_f16(X, A16, ln1w + L * DM, ln1b + L * DM, mA, hi);

    // ---- QKV = xn @ in_proj^T + b -> Bf[16][192] f16 ----
    {
      const float* Wl = inw + (size_t)L * 192 * DM;
      const float* bl = inb + L * 192;
#pragma unroll
      for (int nt = 0; nt < 12; ++nt) {
        const int n0 = nt * 16;
        v8f c = {0.f,0.f,0.f,0.f,0.f,0.f,0.f,0.f};
#pragma unroll
        for (int kb = 0; kb < DM; kb += 32) {
          v16h a = loadA_lds(A16, DM, mA, hi, kb);
          v16h b = loadB_gl(Wl, DM, n0 + nl, hi, kb);
          c = wmma_f16(a, b, c);
        }
        const float bias = bl[n0 + nl];
#pragma unroll
        for (int r = 0; r < 8; ++r)
          Bf[(r + 8 * hi) * 192 + n0 + nl] = (_Float16)(c[r] + bias);
      }
    }

    // ---- attention (tiny 8x8 per head: VALU + LDS), ao -> A16[16][64] f16 ----
    {
      const int h = (l >> 3) & 3;  // each lane owns one (h,q) pair
      const int q = l & 7;
#pragma unroll
      for (int e = 0; e < EPW; ++e) {
        const int mq = 8 * e + q;
        const _Float16* qp = Bf + mq * 192 + h * HD;
        v8h q0 = *(const v8h*)(qp);
        v8h q1 = *(const v8h*)(qp + 8);
        float sc[NTOK];
#pragma unroll
        for (int kt = 0; kt < NTOK; ++kt) {
          const _Float16* kp = Bf + (8 * e + kt) * 192 + DM + h * HD;
          v8h k0 = *(const v8h*)(kp);
          v8h k1 = *(const v8h*)(kp + 8);
          float d = 0.f;
#pragma unroll
          for (int t = 0; t < 8; ++t)
            d += (float)q0[t] * (float)k0[t] + (float)q1[t] * (float)k1[t];
          sc[kt] = d * 0.25f;  // 1/sqrt(HD)
        }
        float mx = sc[0];
#pragma unroll
        for (int kt = 1; kt < NTOK; ++kt) mx = fmaxf(mx, sc[kt]);
        float su = 0.f;
#pragma unroll
        for (int kt = 0; kt < NTOK; ++kt) { sc[kt] = expf(sc[kt] - mx); su += sc[kt]; }
        const float inv = 1.0f / su;
#pragma unroll
        for (int kt = 0; kt < NTOK; ++kt) sc[kt] *= inv;

        if (L == 0) {
#pragma unroll
          for (int kt = 0; kt < NTOK; ++kt) probs0[e][kt] = sc[kt];
        } else {
          float* ap = out_attn + (((size_t)(e0 + e) * NH + h) * NTOK + q) * NTOK;
          v4f o0, o1;
#pragma unroll
          for (int kt = 0; kt < 4; ++kt) {
            o0[kt] = 0.5f * (probs0[e][kt] + sc[kt]);
            o1[kt] = 0.5f * (probs0[e][kt + 4] + sc[kt + 4]);
          }
          *(v4f*)(ap)     = o0;
          *(v4f*)(ap + 4) = o1;
        }

        float acc[HD];
#pragma unroll
        for (int t = 0; t < HD; ++t) acc[t] = 0.f;
#pragma unroll
        for (int kt = 0; kt < NTOK; ++kt) {
          const _Float16* vp = Bf + (8 * e + kt) * 192 + 2 * DM + h * HD;
          v8h v0 = *(const v8h*)(vp);
          v8h v1 = *(const v8h*)(vp + 8);
#pragma unroll
          for (int t = 0; t < 8; ++t) {
            acc[t]     += sc[kt] * (float)v0[t];
            acc[t + 8] += sc[kt] * (float)v1[t];
          }
        }
        v8h h0, h1;
#pragma unroll
        for (int t = 0; t < 8; ++t) { h0[t] = (_Float16)acc[t]; h1[t] = (_Float16)acc[t + 8]; }
        _Float16* op = A16 + mq * DM + h * HD;
        *(v8h*)(op)     = h0;
        *(v8h*)(op + 8) = h1;
      }
    }

    // ---- out_proj + residual -> X ----
    {
      const float* Wl = outw + (size_t)L * DM * DM;
      const float* bl = outb + L * DM;
#pragma unroll
      for (int nt = 0; nt < 4; ++nt) {
        const int n0 = nt * 16;
        v8f c = {0.f,0.f,0.f,0.f,0.f,0.f,0.f,0.f};
#pragma unroll
        for (int kb = 0; kb < DM; kb += 32) {
          v16h a = loadA_lds(A16, DM, mA, hi, kb);
          v16h b = loadB_gl(Wl, DM, n0 + nl, hi, kb);
          c = wmma_f16(a, b, c);
        }
        const float bias = bl[n0 + nl];
#pragma unroll
        for (int r = 0; r < 8; ++r) {
          const int idx = (r + 8 * hi) * DM + n0 + nl;
          X[idx] += c[r] + bias;
        }
      }
    }

    // ---- LN2 -> A16 ----
    ln_to_f16(X, A16, ln2w + L * DM, ln2b + L * DM, mA, hi);

    // ---- FF1 + exact GELU -> Bf[16][256] f16 ----
    {
      const float* Wl = f1w + (size_t)L * DFF * DM;
      const float* bl = f1b + L * DFF;
#pragma unroll
      for (int nt = 0; nt < 16; ++nt) {
        const int n0 = nt * 16;
        v8f c = {0.f,0.f,0.f,0.f,0.f,0.f,0.f,0.f};
#pragma unroll
        for (int kb = 0; kb < DM; kb += 32) {
          v16h a = loadA_lds(A16, DM, mA, hi, kb);
          v16h b = loadB_gl(Wl, DM, n0 + nl, hi, kb);
          c = wmma_f16(a, b, c);
        }
        const float bias = bl[n0 + nl];
#pragma unroll
        for (int r = 0; r < 8; ++r) {
          const float v = c[r] + bias;
          const float g = 0.5f * v * (1.0f + erff(v * 0.70710678118f));
          Bf[(r + 8 * hi) * DFF + n0 + nl] = (_Float16)g;
        }
      }
    }

    // ---- FF2 + residual -> X (K = 256, chained WMMA accumulation) ----
    {
      const float* Wl = f2w + (size_t)L * DM * DFF;
      const float* bl = f2b + L * DM;
#pragma unroll
      for (int nt = 0; nt < 4; ++nt) {
        const int n0 = nt * 16;
        v8f c = {0.f,0.f,0.f,0.f,0.f,0.f,0.f,0.f};
#pragma unroll
        for (int kb = 0; kb < DFF; kb += 32) {
          v16h a = loadA_lds(Bf, DFF, mA, hi, kb);
          v16h b = loadB_gl(Wl, DFF, n0 + nl, hi, kb);
          c = wmma_f16(a, b, c);
        }
        const float bias = bl[n0 + nl];
#pragma unroll
        for (int r = 0; r < 8; ++r) {
          const int idx = (r + 8 * hi) * DM + n0 + nl;
          X[idx] += c[r] + bias;
        }
      }
    }
  }

  // ---- heads: mean-pool over tokens, actor/critic MLPs (tiny -> VALU) ----
  float* P = (float*)A16;  // 512-float scratch: pooled[2][64], a[2][64], c[2][64]
#pragma unroll
  for (int i = 0; i < 4; ++i) {
    const int idx = l + 32 * i;
    const int e = idx >> 6, d = idx & 63;
    float s = 0.f;
#pragma unroll
    for (int t = 0; t < NTOK; ++t) s += X[(8 * e + t) * DM + d];
    P[idx] = s * 0.125f;
  }
#pragma unroll
  for (int i = 0; i < 4; ++i) {
    const int idx = l + 32 * i;
    const int e = idx >> 6, o = idx & 63;
    float aa = a1b[o], cc = c1b[o];
    for (int d = 0; d < DM; ++d) {
      const float p = P[e * DM + d];
      aa += p * a1w[o * DM + d];
      cc += p * c1w[o * DM + d];
    }
    P[128 + idx] = fmaxf(aa, 0.f);
    P[256 + idx] = fmaxf(cc, 0.f);
  }
  if (l < 8) {
    const int e = l >> 2, j = l & 3;
    float acc = a2b[j];
    for (int d = 0; d < DM; ++d) acc += P[128 + e * DM + d] * a2w[j * DM + d];
    out_logits[(size_t)(e0 + e) * 4 + j] = acc;
  } else if (l < 10) {
    const int e = l - 8;
    float acc = c2b[0];
    for (int d = 0; d < DM; ++d) acc += P[256 + e * DM + d] * c2w[d];
    out_value[e0 + e] = acc;
  }
}

extern "C" void kernel_launch(void* const* d_in, const int* in_sizes, int n_in,
                              void* d_out, int out_size, void* d_ws, size_t ws_size,
                              hipStream_t stream) {
  (void)n_in; (void)out_size; (void)d_ws; (void)ws_size;
  const int B = in_sizes[0] / 400;
  float* out        = (float*)d_out;
  float* out_logits = out;                    // (B, 4)
  float* out_value  = out + (size_t)B * 4;    // (B, 1)
  float* out_attn   = out + (size_t)B * 5;    // (B, 4, 8, 8)

  dim3 grid(B / (WAVES * EPW));
  dim3 block(WAVES * 32);
  hipLaunchKernelGGL(ppo_transformer_fwd, grid, block, 0, stream,
      (const float*)d_in[0],  (const float*)d_in[1],  (const float*)d_in[2],
      (const float*)d_in[3],  (const float*)d_in[4],  (const float*)d_in[5],
      (const float*)d_in[6],  (const float*)d_in[7],  (const float*)d_in[8],
      (const float*)d_in[9],  (const float*)d_in[10], (const float*)d_in[11],
      (const float*)d_in[12], (const float*)d_in[13], (const float*)d_in[14],
      (const float*)d_in[15], (const float*)d_in[16], (const float*)d_in[17],
      (const float*)d_in[18], (const float*)d_in[19], (const float*)d_in[20],
      (const float*)d_in[21], (const float*)d_in[22], (const float*)d_in[23],
      out_logits, out_value, out_attn);
}
